// Tpmann_6640019440393
// MI455X (gfx1250) — matrix operand order repaired and checked by reading, not compiled, not run on hardware
//
#include <hip/hip_runtime.h>
#include <math.h>

// ---------------- problem constants (match reference) ----------------
namespace {
constexpr int kB    = 256;
constexpr int kS    = 30;
constexpr int kW    = 10;
constexpr int kSym  = 128;
constexpr int kHid  = 512;
constexpr int kEnt  = 64;
constexpr int kRole = 32;
constexpr int kSpad = 32;                       // S padded to 32 for WMMA K
constexpr int kEntP  = kEnt + 1;                // LDS pad (bank conflicts)
constexpr int kRoleP = kRole + 1;
constexpr int kTpr  = kEnt * kRole * kEnt;      // 131072 per batch
constexpr int kLayers = 3;
}

typedef __attribute__((ext_vector_type(16))) __bf16 v16bf;
typedef __attribute__((ext_vector_type(8)))  __bf16 bf16x8;
typedef __attribute__((ext_vector_type(8)))  float  v8f;

// ---------------- bf16 helpers (native hardware cvt, RNE) ----------------
__device__ __forceinline__ __bf16 f2b(float f) { return (__bf16)f; }
__device__ __forceinline__ unsigned short f2bu(float f) {
  return __builtin_bit_cast(unsigned short, (__bf16)f);
}

__device__ __forceinline__ v16bf cat8(bf16x8 lo, bf16x8 hi) {
  return __builtin_shufflevector(lo, hi, 0, 1, 2, 3, 4, 5, 6, 7,
                                 8, 9, 10, 11, 12, 13, 14, 15);
}

// =====================================================================
// Small utility kernels
// =====================================================================

// dst[N,K] (bf16) = transpose of src[K,N] (f32) -- one-time weight prep so
// the WMMA B fragment becomes two contiguous 8-element runs per lane.
__global__ void cvt_tr_kernel(const float* __restrict__ src,
                              unsigned short* __restrict__ dst, int K, int N) {
  int idx = blockIdx.x * blockDim.x + threadIdx.x;
  if (idx < K * N) {
    int nrow = idx / K, k = idx % K;
    dst[idx] = f2bu(src[(size_t)k * N + nrow]);
  }
}

// sent[b,s,e] = sum_w word_embed[story[b,s,w], e] * pos_embed[w, e]  -> bf16
__global__ void embed_sent_kernel(const int* __restrict__ story,
                                  const float* __restrict__ we,
                                  const float* __restrict__ pos,
                                  unsigned short* __restrict__ sentb) {
  int idx = blockIdx.x * blockDim.x + threadIdx.x;   // over B*S*SYM (exact)
  int e  = idx & (kSym - 1);
  int bs = idx >> 7;
  float acc = 0.f;
#pragma unroll
  for (int w = 0; w < kW; ++w) {
    int tok = story[bs * kW + w];
    acc += we[tok * kSym + e] * pos[w * kSym + e];
  }
  sentb[idx] = f2bu(acc);
}

__global__ void embed_query_kernel(const int* __restrict__ query,
                                   const float* __restrict__ we,
                                   const float* __restrict__ pos,
                                   unsigned short* __restrict__ qb) {
  int idx = blockIdx.x * blockDim.x + threadIdx.x;   // over B*SYM (exact)
  int e = idx & (kSym - 1);
  int b = idx >> 7;
  float acc = 0.f;
#pragma unroll
  for (int w = 0; w < kW; ++w) {
    int tok = query[b * kW + w];
    acc += we[tok * kSym + e] * pos[w * kSym + e];
  }
  qb[idx] = f2bu(acc);
}

// tpr[b, :] = fwm[:] broadcast
__global__ void tpr_init_kernel(float* __restrict__ tpr,
                                const float* __restrict__ fwm) {
  size_t idx = (size_t)blockIdx.x * blockDim.x + threadIdx.x;  // B*kTpr exact
  tpr[idx] = fwm[idx & (size_t)(kTpr - 1)];
}

// =====================================================================
// Generic bf16 WMMA GEMM:  out = tanh(A[M,K] @ W[K,N] + bias[N])
// W supplied pre-transposed as WT[N,K].  One wave per 16x16 output tile.
// A fragment: two b128 loads per lane; B fragment: two b128 loads per lane.
// =====================================================================
__global__ __launch_bounds__(32)
void gemm_bias_tanh_kernel(const unsigned short* __restrict__ A,
                           const unsigned short* __restrict__ WT,
                           const float* __restrict__ bias,
                           unsigned short* __restrict__ outb,  // nullable
                           float* __restrict__ outf,           // nullable
                           int M, int N, int K) {
  const int lane = threadIdx.x;
  const int lh = lane >> 4, ll = lane & 15;
  const int m0 = blockIdx.y * 16, n0 = blockIdx.x * 16;
  const int mA = m0 + ll;      // A row this lane supplies
  const int n  = n0 + ll;      // B/C column this lane owns
  (void)M;
  const unsigned short* Arow = A + (size_t)mA * K;
  const unsigned short* Brow = WT + (size_t)n * K;
  v8f c = {};
  for (int k0 = 0; k0 < K; k0 += 32) {
    // A[16,32]: lane ll holds row mA; elements 0..7 = k0+8*lh.., 8..15 = k0+16+8*lh..
    bf16x8 alo = *(const bf16x8*)(Arow + k0 + 8 * lh);
    bf16x8 ahi = *(const bf16x8*)(Arow + k0 + 16 + 8 * lh);
    // B[32,16]: lane owns column n; elements 0..7 = k0+8*lh.., 8..15 = k0+16+8*lh..
    bf16x8 blo = *(const bf16x8*)(Brow + k0 + 8 * lh);
    bf16x8 bhi = *(const bf16x8*)(Brow + k0 + 16 + 8 * lh);
    c = __builtin_amdgcn_wmma_f32_16x16x32_bf16(false, cat8(alo, ahi),
                                                false, cat8(blo, bhi),
                                                (short)0, c, false, false);
  }
  const float bn = bias[n];
#pragma unroll
  for (int v = 0; v < 8; ++v) {
    int m = m0 + v + 8 * lh;
    float val = tanhf(c[v] + bn);
    if (outf) outf[(size_t)m * N + n] = val;
    if (outb) outb[(size_t)m * N + n] = f2bu(val);
  }
}

// =====================================================================
// TPR solver: one workgroup (8 wave32) per batch, 3 layers of updates.
// tpr master state fp32 in global (512KB/batch, 128MB total -> L2-resident);
// all contractions via bf16 WMMA with fragments built on the fly from LDS.
// =====================================================================

// hat[s,f] = sum_{e,r} eS[s,e]*rS[s,r]*T[e,r,f]
// GEMM: A[32s, 32r] x B[32r, 16f] accumulated over 64 e-blocks.
__device__ __forceinline__ void hat_gemm(const float* __restrict__ T,
                                         const float (*eS)[kEntP],
                                         const float (*rS)[kRoleP],
                                         float (*dst)[kEntP],
                                         int wave, int lane) {
  const int stile = wave >> 2;          // 0..1
  const int ftile = wave & 3;           // 0..3
  const int lh = lane >> 4, ll = lane & 15;
  const int m = stile * 16 + ll;        // s row supplied by this lane (A)
  const int n = ftile * 16 + ll;        // f column owned by this lane (B/C)
  v8f c = {};
  for (int e = 0; e < kEnt; ++e) {
    const float ev = eS[m][e];
    const float* Te = T + (size_t)e * (kRole * kEnt);
    // prefetch next e-slice of T (32 lanes x 64-float stride cover all 8KB)
    if (e + 1 < kEnt)
      __builtin_prefetch(Te + kRole * kEnt + (lane << 6), 0, 3);
    v16bf a, bf;
#pragma unroll
    for (int v = 0; v < 8; ++v) {
      int kk = 16 * (v >> 2) + 8 * lh + 2 * (v & 3);   // r index (A)
      a[2 * v]     = f2b(ev * rS[m][kk]);
      a[2 * v + 1] = f2b(ev * rS[m][kk + 1]);
      int kb = 16 * (v >> 2) + 8 * lh + 2 * (v & 3);   // r index (B)
      bf[2 * v]     = f2b(Te[(size_t)kb * kEnt + n]);
      bf[2 * v + 1] = f2b(Te[(size_t)(kb + 1) * kEnt + n]);
    }
    c = __builtin_amdgcn_wmma_f32_16x16x32_bf16(false, a, false, bf,
                                                (short)0, c, false, false);
  }
#pragma unroll
  for (int v = 0; v < 8; ++v) dst[stile * 16 + v + 8 * lh][n] = c[v];
}

// T[e, r, f] += sum_s eC[s,e] * ( rS[s,r] * (v1[s,f] - v2[s,f]) )
// GEMM per 16x16 tile: A[16e, 32s] x B[32s, 16(rf)], C preloaded from T.
__device__ __forceinline__ void upd_gemm(float* __restrict__ T,
                                         const float (*eC)[kEntP],
                                         const float (*rS)[kRoleP],
                                         const float (*v1)[kEntP],
                                         const float (*v2)[kEntP],
                                         int wave, int lane) {
  const int lh = lane >> 4, ll = lane & 15;
  for (int t = 0; t < 64; ++t) {
    const int id = wave * 64 + t;       // 512 tiles: 4 e-tiles x 128 rf-tiles
    const int etile = id >> 7;
    const int nn = (id & 127) * 16 + ll;        // column in [0,2048)
    const int rcol = nn >> 6, fcol = nn & 63;
    const int me = etile * 16 + ll;             // e row supplied by this lane
    v16bf a, bf;
#pragma unroll
    for (int v = 0; v < 8; ++v) {
      int ks = 16 * (v >> 2) + 8 * lh + 2 * (v & 3);   // s index (A)
      a[2 * v]     = f2b(eC[ks][me]);
      a[2 * v + 1] = f2b(eC[ks + 1][me]);
      int kb = 16 * (v >> 2) + 8 * lh + 2 * (v & 3);   // s index (B)
      bf[2 * v]     = f2b(rS[kb][rcol] * (v1[kb][fcol] - v2[kb][fcol]));
      bf[2 * v + 1] = f2b(rS[kb + 1][rcol] * (v1[kb + 1][fcol] - v2[kb + 1][fcol]));
    }
    const size_t base = (size_t)(etile * 16) * (kRole * kEnt) + nn;
    v8f c;
#pragma unroll
    for (int v = 0; v < 8; ++v)
      c[v] = T[base + (size_t)(v + 8 * lh) * (kRole * kEnt)];
    c = __builtin_amdgcn_wmma_f32_16x16x32_bf16(false, a, false, bf,
                                                (short)0, c, false, false);
#pragma unroll
    for (int v = 0; v < 8; ++v)
      T[base + (size_t)(v + 8 * lh) * (kRole * kEnt)] = c[v];
  }
}

__global__ __launch_bounds__(256)
void tpr_solver_kernel(float* __restrict__ tpr,
                       const float* __restrict__ E1, const float* __restrict__ E2,
                       const float* __restrict__ R1, const float* __restrict__ R2,
                       const float* __restrict__ R3) {
  __shared__ float se1[kSpad][kEntP];
  __shared__ float se2[kSpad][kEntP];
  __shared__ float sr1[kSpad][kRoleP];
  __shared__ float sr2[kSpad][kRoleP];
  __shared__ float sr3[kSpad][kRoleP];
  __shared__ float hatA[kSpad][kEntP];
  __shared__ float hatB[kSpad][kEntP];
  __shared__ float red[256];

  const int b   = blockIdx.x;
  const int tid = threadIdx.x;
  float* T = tpr + (size_t)b * kTpr;

  for (int i = tid; i < kSpad * kEnt; i += 256) {
    int s = i / kEnt, f = i % kEnt;
    float a = 0.f, bb = 0.f;
    if (s < kS) {
      a  = E1[((size_t)b * kS + s) * kEnt + f];
      bb = E2[((size_t)b * kS + s) * kEnt + f];
    }
    se1[s][f] = a; se2[s][f] = bb;
  }
  for (int i = tid; i < kSpad * kRole; i += 256) {
    int s = i / kRole, r = i % kRole;
    float a = 0.f, bb = 0.f, cc = 0.f;
    if (s < kS) {
      a  = R1[((size_t)b * kS + s) * kRole + r];
      bb = R2[((size_t)b * kS + s) * kRole + r];
      cc = R3[((size_t)b * kS + s) * kRole + r];
    }
    sr1[s][r] = a; sr2[s][r] = bb; sr3[s][r] = cc;
  }
  __syncthreads();

  const int lane = tid & 31;
  const int wave = tid >> 5;

  for (int layer = 0; layer < kLayers; ++layer) {
    // w_hat = <e1, r1, T>
    hat_gemm(T, se1, sr1, hatA, wave, lane);
    __syncthreads();
    // T += e1^T (r1 * (e2 - w_hat))
    upd_gemm(T, se1, sr1, se2, hatA, wave, lane);
    __threadfence(); __syncthreads();
    // m_hat = <e1, r2, T>
    hat_gemm(T, se1, sr2, hatB, wave, lane);
    __syncthreads();
    // T += e1^T (r2 * (w_hat - m_hat))
    upd_gemm(T, se1, sr2, hatA, hatB, wave, lane);
    __threadfence(); __syncthreads();
    // b_hat = <e2, r3, T>
    hat_gemm(T, se2, sr3, hatA, wave, lane);
    __syncthreads();
    // T += e2^T (r3 * (e1 - b_hat))
    upd_gemm(T, se2, sr3, se1, hatA, wave, lane);
    __threadfence(); __syncthreads();

    // Frobenius norm clip: T /= max(||T||_F, 1)
    float acc = 0.f;
    for (int i = tid; i < kTpr; i += 256) { float v = T[i]; acc += v * v; }
    red[tid] = acc;
    __syncthreads();
    for (int s = 128; s > 0; s >>= 1) {
      if (tid < s) red[tid] += red[tid + s];
      __syncthreads();
    }
    if (tid == 0) {
      float fn = sqrtf(red[0]);
      red[0] = 1.f / (fn < 1.f ? 1.f : fn);
    }
    __syncthreads();
    const float sc = red[0];
    __syncthreads();
    for (int i = tid; i < kTpr; i += 256) T[i] *= sc;
    __threadfence(); __syncthreads();
  }
}

// =====================================================================
// Inference: 3 chained contractions + layernorm (+ final @Z), 1 block/batch
// =====================================================================
__global__ __launch_bounds__(64)
void infer_kernel(const float* __restrict__ tpr,
                  const float* __restrict__ qE1,
                  const float* __restrict__ qR1,
                  const float* __restrict__ qR2,
                  const float* __restrict__ qR3,
                  const float* __restrict__ ln_gain,
                  const float* __restrict__ ln_bias,
                  const float* __restrict__ Zm,
                  float* __restrict__ out) {
  __shared__ float vprev[kEnt];
  __shared__ float rv[kRole];
  __shared__ float red[kEnt];
  __shared__ float isum[kEnt];
  const int b = blockIdx.x, f = threadIdx.x;
  const float* T = tpr + (size_t)b * kTpr;

  vprev[f] = qE1[b * kEnt + f];
  isum[f] = 0.f;
  __syncthreads();

  for (int st = 0; st < 3; ++st) {
    const float* qr = (st == 0) ? qR1 : (st == 1) ? qR2 : qR3;
    if (f < kRole) rv[f] = qr[b * kRole + f];
    __syncthreads();
    float acc = 0.f;
    for (int e = 0; e < kEnt; ++e) {
      const float ce = vprev[e];
      const float* Te = T + (size_t)e * (kRole * kEnt);
#pragma unroll 4
      for (int r = 0; r < kRole; ++r) acc += ce * rv[r] * Te[r * kEnt + f];
    }
    // layernorm over f (ddof=1)
    red[f] = acc; __syncthreads();
    for (int s = 32; s > 0; s >>= 1) { if (f < s) red[f] += red[f + s]; __syncthreads(); }
    const float mean = red[0] / (float)kEnt;
    __syncthreads();
    const float d = acc - mean;
    red[f] = d * d; __syncthreads();
    for (int s = 32; s > 0; s >>= 1) { if (f < s) red[f] += red[f + s]; __syncthreads(); }
    const float sd = sqrtf(red[0] / (float)(kEnt - 1));
    const float y = ln_gain[st * kEnt + f] * d / (sd + 1e-6f) + ln_bias[st * kEnt + f];
    isum[f] += y;
    __syncthreads();            // everyone past the vprev reads (tree barriers above)
    vprev[f] = y;
    __syncthreads();
  }
  if (f < 9) {
    float a = 0.f;
#pragma unroll
    for (int ff = 0; ff < kEnt; ++ff) a += isum[ff] * Zm[ff * 9 + f];
    out[b * 9 + f] = a;
  }
}

// =====================================================================
// Host orchestration
// =====================================================================
extern "C" void kernel_launch(void* const* d_in, const int* in_sizes, int n_in,
                              void* d_out, int out_size, void* d_ws, size_t ws_size,
                              hipStream_t stream) {
  (void)in_sizes; (void)n_in; (void)out_size; (void)ws_size;
  const int*   story      = (const int*)d_in[0];
  const int*   query      = (const int*)d_in[1];
  const float* word_embed = (const float*)d_in[2];
  const float* pos_embed  = (const float*)d_in[3];
  const float* ue_W1 = (const float*)d_in[4];
  const float* ue_b1 = (const float*)d_in[5];
  const float* ue_W2 = (const float*)d_in[6];
  const float* ue_b2 = (const float*)d_in[7];
  const float* ur_W1 = (const float*)d_in[8];
  const float* ur_b1 = (const float*)d_in[9];
  const float* ur_W2 = (const float*)d_in[10];
  const float* ur_b2 = (const float*)d_in[11];
  const float* fwm   = (const float*)d_in[12];
  const float* ie_W1 = (const float*)d_in[13];
  const float* ie_b1 = (const float*)d_in[14];
  const float* ie_W2 = (const float*)d_in[15];
  const float* ie_b2 = (const float*)d_in[16];
  const float* ir_W1 = (const float*)d_in[17];
  const float* ir_b1 = (const float*)d_in[18];
  const float* ir_W2 = (const float*)d_in[19];
  const float* ir_b2 = (const float*)d_in[20];
  const float* ln_gain = (const float*)d_in[21];
  const float* ln_bias = (const float*)d_in[22];
  const float* Zm      = (const float*)d_in[23];
  float* out = (float*)d_out;

  char* wsb = (char*)d_ws;
  size_t off = 0;
  auto alloc = [&](size_t bytes) -> void* {
    off = (off + 255) & ~(size_t)255;
    void* p = wsb + off;
    off += bytes;
    return p;
  };
  float*          tpr   = (float*)alloc((size_t)kB * kTpr * 4);          // 128 MB (L2-resident)
  unsigned short* sentb = (unsigned short*)alloc((size_t)kB * kS * kSym * 2);
  unsigned short* qb    = (unsigned short*)alloc((size_t)kB * kSym * 2);
  unsigned short* hid   = (unsigned short*)alloc((size_t)kB * kS * kHid * 2);
  float* E1 = (float*)alloc((size_t)kB * kS * kEnt * 4);
  float* E2 = (float*)alloc((size_t)kB * kS * kEnt * 4);
  float* R1 = (float*)alloc((size_t)kB * kS * kRole * 4);
  float* R2 = (float*)alloc((size_t)kB * kS * kRole * 4);
  float* R3 = (float*)alloc((size_t)kB * kS * kRole * 4);
  float* qE1 = (float*)alloc((size_t)kB * kEnt * 4);
  float* qR1 = (float*)alloc((size_t)kB * kRole * 4);
  float* qR2 = (float*)alloc((size_t)kB * kRole * 4);
  float* qR3 = (float*)alloc((size_t)kB * kRole * 4);
  // weights, pre-transposed to [N,K] bf16
  unsigned short* ueW1t = (unsigned short*)alloc((size_t)2 * kSym * kHid * 2);
  unsigned short* ueW2t = (unsigned short*)alloc((size_t)2 * kHid * kEnt * 2);
  unsigned short* urW1t = (unsigned short*)alloc((size_t)3 * kSym * kHid * 2);
  unsigned short* urW2t = (unsigned short*)alloc((size_t)3 * kHid * kRole * 2);
  unsigned short* ieW1t = (unsigned short*)alloc((size_t)kSym * kHid * 2);   // head 0 only used
  unsigned short* ieW2t = (unsigned short*)alloc((size_t)kHid * kEnt * 2);
  unsigned short* irW1t = (unsigned short*)alloc((size_t)3 * kSym * kHid * 2);
  unsigned short* irW2t = (unsigned short*)alloc((size_t)3 * kHid * kRole * 2);

  auto cvtT = [&](const float* src, unsigned short* dst, int K, int N) {
    int n = K * N;
    cvt_tr_kernel<<<(n + 255) / 256, 256, 0, stream>>>(src, dst, K, N);
  };
  // per-head transposed bf16 weights
  for (int h = 0; h < 2; ++h) {
    cvtT(ue_W1 + (size_t)h * kSym * kHid, ueW1t + (size_t)h * kSym * kHid, kSym, kHid);
    cvtT(ue_W2 + (size_t)h * kHid * kEnt, ueW2t + (size_t)h * kHid * kEnt, kHid, kEnt);
  }
  for (int h = 0; h < 3; ++h) {
    cvtT(ur_W1 + (size_t)h * kSym * kHid,  urW1t + (size_t)h * kSym * kHid,  kSym, kHid);
    cvtT(ur_W2 + (size_t)h * kHid * kRole, urW2t + (size_t)h * kHid * kRole, kHid, kRole);
    cvtT(ir_W1 + (size_t)h * kSym * kHid,  irW1t + (size_t)h * kSym * kHid,  kSym, kHid);
    cvtT(ir_W2 + (size_t)h * kHid * kRole, irW2t + (size_t)h * kHid * kRole, kHid, kRole);
  }
  cvtT(ie_W1, ieW1t, kSym, kHid);
  cvtT(ie_W2, ieW2t, kHid, kEnt);

  embed_sent_kernel<<<(kB * kS * kSym) / 256, 256, 0, stream>>>(story, word_embed, pos_embed, sentb);
  embed_query_kernel<<<(kB * kSym) / 256, 256, 0, stream>>>(query, word_embed, pos_embed, qb);
  tpr_init_kernel<<<(kB * kTpr) / 256, 256, 0, stream>>>(tpr, fwm);

  auto gemm = [&](const unsigned short* A, const unsigned short* Wt, const float* bias,
                  unsigned short* ob, float* of, int M, int N, int K) {
    dim3 g(N / 16, M / 16);
    gemm_bias_tanh_kernel<<<g, 32, 0, stream>>>(A, Wt, bias, ob, of, M, N, K);
  };
  const int MS = kB * kS;  // 7680 (= 480 * 16)
  // entity heads: e1, e2
  gemm(sentb, ueW1t + 0 * kSym * kHid, ue_b1 + 0 * kHid, hid, nullptr, MS, kHid, kSym);
  gemm(hid,   ueW2t + 0 * kHid * kEnt, ue_b2 + 0 * kEnt, nullptr, E1,  MS, kEnt, kHid);
  gemm(sentb, ueW1t + 1 * kSym * kHid, ue_b1 + 1 * kHid, hid, nullptr, MS, kHid, kSym);
  gemm(hid,   ueW2t + 1 * kHid * kEnt, ue_b2 + 1 * kEnt, nullptr, E2,  MS, kEnt, kHid);
  // role heads: r1, r2, r3
  float* Rp[3] = {R1, R2, R3};
  for (int n = 0; n < 3; ++n) {
    gemm(sentb, urW1t + n * kSym * kHid,  ur_b1 + n * kHid,  hid, nullptr, MS, kHid,  kSym);
    gemm(hid,   urW2t + n * kHid * kRole, ur_b2 + n * kRole, nullptr, Rp[n], MS, kRole, kHid);
  }
  // query heads (only qe1 and qr1..3 are used by the reference)
  gemm(qb,  ieW1t, ie_b1, hid, nullptr, kB, kHid, kSym);
  gemm(hid, ieW2t, ie_b2, nullptr, qE1, kB, kEnt, kHid);
  float* qRp[3] = {qR1, qR2, qR3};
  for (int n = 0; n < 3; ++n) {
    gemm(qb,  irW1t + n * kSym * kHid,  ir_b1 + n * kHid,  hid, nullptr, kB, kHid,  kSym);
    gemm(hid, irW2t + n * kHid * kRole, ir_b2 + n * kRole, nullptr, qRp[n], kB, kRole, kHid);
  }

  // 3-layer TPR update: one workgroup per batch, bf16 WMMA throughout
  tpr_solver_kernel<<<kB, 256, 0, stream>>>(tpr, E1, E2, R1, R2, R3);

  // inference + layernorm + output projection
  infer_kernel<<<kB, 64, 0, stream>>>(tpr, qE1, qR1, qR2, qR3, ln_gain, ln_bias, Zm, out);
}
